// ResCo_52716428591444
// MI455X (gfx1250) — matrix operand order repaired and checked by reading, not compiled, not run on hardware
//
#include <hip/hip_runtime.h>
#include <hip/hip_bf16.h>

typedef __bf16 bf16;
typedef __attribute__((ext_vector_type(16))) __bf16 v16bf;
typedef __attribute__((ext_vector_type(8)))  float  v8f;
typedef __attribute__((ext_vector_type(4)))  int    v4i;

#define Bq   1024
#define DFv  2048
#define DIMv 128
#define Cv   8192
#define Pv   8
#define Nv   4
#define NEGW (Nv*Cv)                      // 32768
#define POSW (Pv*Cv)                      // 65536
#define SIMW (Bq + Pv + Nv*(Cv-1))        // 33796

// ---- output layout (flat f32, reference tuple order) ----
static constexpr size_t O_lab  = (size_t)Bq*SIMW;
static constexpr size_t O_logq = O_lab*2;
static constexpr size_t O_logk = O_logq + (size_t)Bq*Cv;
static constexpr size_t O_nq   = O_logk + (size_t)Bq*Cv;
static constexpr size_t O_pq   = O_nq + (size_t)DIMv*NEGW;
static constexpr size_t O_nptr = O_pq + (size_t)DIMv*POSW;

__device__ __forceinline__ bf16 tobf(float x) { return (bf16)x; }

// K index of vector element i, lane half in this code's A/B fragments (contiguous k = 16*half + i)
__device__ __forceinline__ int ka_idx(int i, int half) {
  return half*16 + i;
}

// ---------------- async global->LDS copy (CDNA5 path, guarded) ----------------
#if __has_builtin(__builtin_amdgcn_global_load_async_to_lds_b128) && \
    __has_builtin(__builtin_amdgcn_s_wait_asynccnt)
#define HAVE_ASYNC_LDS 1
typedef __attribute__((address_space(1))) v4i g_v4i;   // global (AS1) int4
typedef __attribute__((address_space(3))) v4i l_v4i;   // LDS (AS3) int4
__device__ __forceinline__ void cp_async16(const void* g, void* l) {
  __builtin_amdgcn_global_load_async_to_lds_b128((g_v4i*)g, (l_v4i*)l, 0, 0);
}
__device__ __forceinline__ void cp_wait() { __builtin_amdgcn_s_wait_asynccnt(0); }
#else
#define HAVE_ASYNC_LDS 0
__device__ __forceinline__ void cp_async16(const void* g, void* l) {
  *(int4*)l = *(const int4*)g;
}
__device__ __forceinline__ void cp_wait() {}
#endif

// ---------------- packing kernels ----------------
// A panels: frag addr = ((mt*KT + kt)*32 + lane)*16 ; row m = mt*16 + lane%16
__global__ void pack_a_k(bf16* __restrict__ dst, const float* __restrict__ s0,
                         const float* __restrict__ s1, int rows0, int Mt, int KT, int ldk) {
  int tid = blockIdx.x*blockDim.x + threadIdx.x;
  int tile = tid >> 5, lane = tid & 31;
  if (tile >= Mt*KT) return;
  int mt = tile / KT, kt = tile % KT;
  int m = mt*16 + (lane & 15), half = lane >> 4;
  const float* row = (m < rows0) ? (s0 + (size_t)m*ldk) : (s1 + (size_t)(m - rows0)*ldk);
  bf16* d = dst + ((size_t)tile*32 + lane)*16;
#pragma unroll
  for (int i = 0; i < 16; ++i) d[i] = tobf(row[kt*32 + ka_idx(i, half)]);
}

// B panels from K x N row-major: frag addr = ((nt*KT + kt)*32 + lane)*16
__global__ void pack_b_k(bf16* __restrict__ dst, const float* __restrict__ src,
                         int KT, int Nt, int ldn) {
  int tid = blockIdx.x*blockDim.x + threadIdx.x;
  int tile = tid >> 5, lane = tid & 31;
  if (tile >= Nt*KT) return;
  int nt = tile / KT, kt = tile % KT;
  int n = nt*16 + (lane & 15), half = lane >> 4;
  bf16* d = dst + ((size_t)tile*32 + lane)*16;
#pragma unroll
  for (int i = 0; i < 16; ++i) {
    int k = kt*32 + half*16 + i;
    d[i] = tobf(src[(size_t)k*ldn + n]);
  }
}

// B panels from N x K row-major source (i.e. B = src^T), for feat_k^T
__global__ void pack_bt_k(bf16* __restrict__ dst, const float* __restrict__ src,
                          int KT, int Nt, int ldk) {
  int tid = blockIdx.x*blockDim.x + threadIdx.x;
  int tile = tid >> 5, lane = tid & 31;
  if (tile >= Nt*KT) return;
  int nt = tile / KT, kt = tile % KT;
  int n = nt*16 + (lane & 15), half = lane >> 4;
  bf16* d = dst + ((size_t)tile*32 + lane)*16;
#pragma unroll
  for (int i = 0; i < 16; ++i) {
    int k = kt*32 + half*16 + i;
    d[i] = tobf(src[(size_t)n*ldk + k]);
  }
}

// ------ WMMA GEMM, LDS-staged + async double buffer; 128x64 macro-tile per block ------
enum { EPI_H = 0, EPI_FEAT = 1, EPI_LOGIT = 2, EPI_SIM = 3, EPI_NEG = 4 };

template <int MODE>
__launch_bounds__(256)
__global__ void gemm_k(const bf16* __restrict__ Ap, const bf16* __restrict__ Bp,
                       int Mt, int KT, int Ngt, float* __restrict__ out,
                       const float* __restrict__ bias, const int* __restrict__ lab) {
  __shared__ __align__(128) bf16 sA[2][8*512];   // 8 A-tiles (16x32) per buffer
  __shared__ __align__(128) bf16 sB[2][4*512];   // 4 B-tiles (32x16) per buffer
  int Mg = Mt >> 3;
  int mg = blockIdx.x % Mg, ntg = blockIdx.x / Mg;
  int mtBase = mg*8;
  int tid = threadIdx.x, w = tid >> 5, lane = tid & 31;

  auto stage = [&](int buf, int kt) {
#pragma unroll
    for (int c = 0; c < 2; ++c) {                 // A: 8 KB = 512 x 16B
      int idx = tid + c*256;
      int chunk = idx >> 6;                       // which of 8 A-tiles
      int off = (idx & 63)*8;                     // bf16 elems
      cp_async16(Ap + ((size_t)(mtBase + chunk)*KT + kt)*512 + off,
                 &sA[buf][chunk*512 + off]);
    }
    {                                             // B: 4 KB = 256 x 16B
      int chunk = tid >> 6;                       // which of 4 B-tiles
      int off = (tid & 63)*8;
      cp_async16(Bp + ((size_t)(ntg*4 + chunk)*KT + kt)*512 + off,
                 &sB[buf][chunk*512 + off]);
    }
  };

  v8f acc[4] = {{}, {}, {}, {}};
  stage(0, 0);
  int buf = 0;
  for (int kt = 0; kt < KT; ++kt) {
    cp_wait();
    __syncthreads();                              // staged data visible to all waves
    if (kt + 1 < KT) stage(buf ^ 1, kt + 1);      // overlap next copies with WMMA
    v16bf a = *(const v16bf*)(&sA[buf][w*512 + lane*16]);
#pragma unroll
    for (int j = 0; j < 4; ++j) {
      v16bf b = *(const v16bf*)(&sB[buf][j*512 + lane*16]);
      acc[j] = __builtin_amdgcn_wmma_f32_16x16x32_bf16(false, a, false, b,
                                                       (short)0, acc[j], false, false);
    }
    __syncthreads();                              // done reading buf before it is reused
    buf ^= 1;
  }

  int mt = mtBase + w;
  int half = lane >> 4, n0 = lane & 15;
#pragma unroll
  for (int j = 0; j < 4; ++j) {
#pragma unroll
    for (int i = 0; i < 8; ++i) {
      int r = mt*16 + i + 8*half;
      int c = (ntg*4 + j)*16 + n0;
      float v = acc[j][i];
      if (MODE == EPI_H) {
        out[(size_t)r*DFv + c] = v + bias[c];
      } else if (MODE == EPI_FEAT) {
        out[(size_t)r*DIMv + c] = v + bias[c];
      } else if (MODE == EPI_LOGIT) {
        v += bias[c];
        if (r < Bq) out[O_logq + (size_t)r*Cv + c] = v;
        else        out[O_logk + (size_t)(r - Bq)*Cv + c] = v;
      } else if (MODE == EPI_SIM) {
        out[(size_t)r*SIMW + c] = v;
      } else {  // EPI_NEG: drop own-class band of 4 columns, shift the rest
        int band = lab[r]*Nv;
        if (c >= band && c < band + Nv) continue;
        int jj = (c >= band + Nv) ? (c - Nv) : c;
        out[(size_t)r*SIMW + (Bq + Pv) + jj] = v;
      }
    }
  }
}

// ---------------- batchnorm stats (per column, q and k halves separately) ----------------
__global__ void bn_stats_k(const float* __restrict__ H, float* __restrict__ mu,
                           float* __restrict__ rv) {
  int id = blockIdx.x*blockDim.x + threadIdx.x;
  if (id >= 2*DFv) return;
  int col = id & (DFv - 1), hb = id >> 11;
  const float* p = H + (size_t)hb*Bq*DFv + col;
  float s = 0.f, ss = 0.f;
  for (int r = 0; r < Bq; ++r) { float v = p[(size_t)r*DFv]; s += v; ss += v*v; }
  float m = s*(1.f/Bq);
  float var = ss*(1.f/Bq) - m*m;
  mu[id] = m;
  rv[id] = rsqrtf(var + 1e-5f);
}

// BN apply + ReLU + repack into bf16 A-panels
__global__ void bn_pack_k(bf16* __restrict__ dst, const float* __restrict__ H,
                          const float* __restrict__ mu, const float* __restrict__ rv,
                          const float* __restrict__ gamma, const float* __restrict__ beta) {
  const int KT = DFv/32;
  int tid = blockIdx.x*blockDim.x + threadIdx.x;
  int tile = tid >> 5, lane = tid & 31;
  if (tile >= (2*Bq/16)*KT) return;
  int mt = tile / KT, kt = tile % KT;
  int m = mt*16 + (lane & 15), half = lane >> 4;
  int hb = (m >= Bq) ? 1 : 0;
  const float* row = H + (size_t)m*DFv;
  bf16* d = dst + ((size_t)tile*32 + lane)*16;
#pragma unroll
  for (int i = 0; i < 16; ++i) {
    int k = kt*32 + ka_idx(i, half);
    float v = (row[k] - mu[hb*DFv + k])*rv[hb*DFv + k]*gamma[k] + beta[k];
    d[i] = tobf(fmaxf(v, 0.f));
  }
}

// ---------------- row l2 norm (one wave32 per row) ----------------
__global__ void l2norm_k(const float* __restrict__ F, float* __restrict__ Fn) {
  int row = blockIdx.x*(blockDim.x >> 5) + (threadIdx.x >> 5);
  int lane = threadIdx.x & 31;
  if (row >= 2*Bq) return;
  float x[4], ss = 0.f;
#pragma unroll
  for (int t = 0; t < 4; ++t) { x[t] = F[(size_t)row*DIMv + lane + t*32]; ss += x[t]*x[t]; }
#pragma unroll
  for (int o = 16; o > 0; o >>= 1) ss += __shfl_xor(ss, o, 32);
  float rn = 1.f / fmaxf(sqrtf(ss), 1e-12f);
#pragma unroll
  for (int t = 0; t < 4; ++t) Fn[(size_t)row*DIMv + lane + t*32] = x[t]*rn;
}

// ---------------- pos_list: warp p computes dot(feat_q[b], pos_queue[:, lq*P+p]) ----------------
__global__ void pos_list_k(const float* __restrict__ Fn, const float* __restrict__ posq,
                           const int* __restrict__ lq, float* __restrict__ out) {
  int b = blockIdx.x, p = threadIdx.x >> 5, lane = threadIdx.x & 31;
  int c = lq[b];
  float s = 0.f;
#pragma unroll
  for (int t = 0; t < 4; ++t) {
    int d = lane + t*32;
    s += Fn[(size_t)b*DIMv + d] * posq[(size_t)d*POSW + (size_t)c*Pv + p];
  }
#pragma unroll
  for (int o = 16; o > 0; o >>= 1) s += __shfl_xor(s, o, 32);
  if (lane == 0) out[(size_t)b*SIMW + Bq + p] = s;
}

// ---------------- labels_con fill ----------------
__global__ void labels_con_k(const int* __restrict__ lq, const int* __restrict__ lk,
                             float* __restrict__ out) {
  size_t total = (size_t)Bq*SIMW;
  for (size_t idx = (size_t)blockIdx.x*blockDim.x + threadIdx.x; idx < total;
       idx += (size_t)gridDim.x*blockDim.x) {
    int b = (int)(idx / SIMW), c = (int)(idx % SIMW);
    float v;
    if (c < Bq)           v = (lq[b] == lk[c]) ? 1.f : 0.f;
    else if (c < Bq + Pv) v = 1.f;
    else                  v = 0.f;
    out[O_lab + idx] = v;
  }
}

// ---------------- sequential queue update (matches lax.scan order) ----------------
__global__ void queue_update_k(const float* __restrict__ Fn, const int* __restrict__ lk,
                               int* __restrict__ nptr, int* __restrict__ pptr,
                               float* __restrict__ onq, float* __restrict__ opq) {
  __shared__ int sc, snp, spp;
  int d = threadIdx.x;
  for (int b = 0; b < Bq; ++b) {
    if (d == 0) {
      int c = lk[b];
      sc = c;
      snp = nptr[c]; nptr[c] = (snp + 1) & (Nv - 1);
      spp = pptr[c]; pptr[c] = (spp + 1) & (Pv - 1);
    }
    __syncthreads();
    float v = Fn[(size_t)(Bq + b)*DIMv + d];
    onq[(size_t)d*NEGW + (size_t)sc*Nv + snp] = v;
    opq[(size_t)d*POSW + (size_t)sc*Pv + spp] = v;
    __syncthreads();
  }
}

__global__ void ptr_out_k(const int* __restrict__ p, float* __restrict__ out) {
  int id = blockIdx.x*blockDim.x + threadIdx.x;
  if (id < 2*Cv) out[O_nptr + id] = (float)p[id];
}

// ---------------- launch ----------------
extern "C" void kernel_launch(void* const* d_in, const int* in_sizes, int n_in,
                              void* d_out, int out_size, void* d_ws, size_t ws_size,
                              hipStream_t stream) {
  const float* q     = (const float*)d_in[0];
  const float* kk    = (const float*)d_in[1];
  const int*   lq    = (const int*)  d_in[2];
  const int*   lk    = (const int*)  d_in[3];
  const float* W1    = (const float*)d_in[4];
  const float* b1    = (const float*)d_in[5];
  const float* gamma = (const float*)d_in[6];
  const float* beta  = (const float*)d_in[7];
  const float* W2    = (const float*)d_in[8];
  const float* b2    = (const float*)d_in[9];
  const float* Wlin  = (const float*)d_in[10];
  const float* blin  = (const float*)d_in[11];
  const float* negq  = (const float*)d_in[12];
  const float* posq  = (const float*)d_in[13];
  float* out = (float*)d_out;

  // workspace layout
  bf16* Xp    = (bf16*)d_ws;                         // 2048x2048
  bf16* W1p   = Xp    + (size_t)DFv*DFv;             // 2048x2048
  bf16* Wlinp = W1p   + (size_t)DFv*DFv;             // 2048x8192
  bf16* W2p   = Wlinp + (size_t)DFv*Cv;              // 2048x128
  bf16* Hp    = W2p   + (size_t)DFv*DIMv;            // 2048x2048
  bf16* Featp = Hp    + (size_t)DFv*DFv;             // 1024x128
  bf16* FkTp  = Featp + (size_t)Bq*DIMv;             // 128x1024
  bf16* negqp = FkTp  + (size_t)DIMv*Bq;             // 128x32768
  float* Hf   = (float*)(negqp + (size_t)DIMv*NEGW); // 2048x2048 f32
  float* muv  = Hf   + (size_t)2*Bq*DFv;             // 2x2048
  float* rvv  = muv  + 2*DFv;                        // 2x2048
  float* Feat = rvv  + 2*DFv;                        // 2048x128
  float* Fn   = Feat + (size_t)2*Bq*DIMv;            // 2048x128
  int*   nptrw = (int*)(Fn + (size_t)2*Bq*DIMv);     // 8192
  int*   pptrw = nptrw + Cv;                         // 8192

  auto cdiv = [](long a, long b) { return (int)((a + b - 1)/b); };

  // ---- pack operands to bf16 WMMA panels ----
  pack_a_k<<<cdiv((long)128*64*32, 256), 256, 0, stream>>>(Xp, q, kk, Bq, 128, 64, DFv);
  pack_b_k<<<cdiv((long)128*64*32, 256), 256, 0, stream>>>(W1p, W1, 64, 128, DFv);
  pack_b_k<<<cdiv((long)512*64*32, 256), 256, 0, stream>>>(Wlinp, Wlin, 64, 512, Cv);
  pack_b_k<<<cdiv((long)8*64*32,   256), 256, 0, stream>>>(W2p, W2, 64, 8, DIMv);
  pack_b_k<<<cdiv((long)2048*4*32, 256), 256, 0, stream>>>(negqp, negq, 4, 2048, NEGW);

  // ---- H = X@W1 + b1 ----  (blocks = (Mt/8) * Ngt, exact cover)
  gemm_k<EPI_H><<<16*32, 256, 0, stream>>>(Xp, W1p, 128, 64, 32, Hf, b1, nullptr);
  // ---- BN stats + BN/ReLU repack ----
  bn_stats_k<<<cdiv(2*DFv, 256), 256, 0, stream>>>(Hf, muv, rvv);
  bn_pack_k<<<cdiv((long)128*64*32, 256), 256, 0, stream>>>(Hp, Hf, muv, rvv, gamma, beta);
  // ---- Feat = Hbn@W2 + b2 ----
  gemm_k<EPI_FEAT><<<16*2, 256, 0, stream>>>(Hp, W2p, 128, 64, 2, Feat, b2, nullptr);
  // ---- l2 normalize rows, pack feat_q (A) and feat_k^T (B) ----
  l2norm_k<<<256, 256, 0, stream>>>(Feat, Fn);
  pack_a_k<<<cdiv((long)64*4*32, 256), 256, 0, stream>>>(Featp, Fn, Fn, 4096, 64, 4, DIMv);
  pack_bt_k<<<cdiv((long)64*4*32, 256), 256, 0, stream>>>(FkTp, Fn + (size_t)Bq*DIMv, 4, 64, DIMv);

  // ---- sim_batch, pos_list, neg_list (scattered), logits ----
  gemm_k<EPI_SIM><<<8*16, 256, 0, stream>>>(Featp, FkTp, 64, 4, 16, out, nullptr, nullptr);
  pos_list_k<<<Bq, 256, 0, stream>>>(Fn, posq, lq, out);
  gemm_k<EPI_NEG><<<8*512, 256, 0, stream>>>(Featp, negqp, 64, 4, 512, out, nullptr, lq);
  gemm_k<EPI_LOGIT><<<16*128, 256, 0, stream>>>(Xp, Wlinp, 128, 64, 128, out, blin, nullptr);
  labels_con_k<<<8192, 256, 0, stream>>>(lq, lk, out);

  // ---- queues: copy originals then sequential update ----
  (void)hipMemcpyAsync(out + O_nq, (void*)negq, (size_t)DIMv*NEGW*sizeof(float),
                       hipMemcpyDeviceToDevice, stream);
  (void)hipMemcpyAsync(out + O_pq, (void*)posq, (size_t)DIMv*POSW*sizeof(float),
                       hipMemcpyDeviceToDevice, stream);
  (void)hipMemcpyAsync(nptrw, d_in[14], Cv*sizeof(int), hipMemcpyDeviceToDevice, stream);
  (void)hipMemcpyAsync(pptrw, d_in[15], Cv*sizeof(int), hipMemcpyDeviceToDevice, stream);
  queue_update_k<<<1, DIMv, 0, stream>>>(Fn, lk, nptrw, pptrw, out + O_nq, out + O_pq);
  ptr_out_k<<<cdiv(2*Cv, 256), 256, 0, stream>>>(nptrw, out);
}
